// Gnn4d_30614526886009
// MI455X (gfx1250) — matrix operand order repaired and checked by reading, not compiled, not run on hardware
//
#include <hip/hip_runtime.h>
#include <stdint.h>

// ---------------------------------------------------------------------------
// Problem constants (from the reference)
// ---------------------------------------------------------------------------
#define BB   4
#define LL   64
#define NN   256          // NUM_S * S
#define QQ   1024
#define OUTC 64
#define COLS (NN * QQ)    // 262144 columns of the per-batch (l x (m,q)) matrix

typedef __attribute__((ext_vector_type(16))) __bf16 bf16x16;
typedef __attribute__((ext_vector_type(8)))  float  f32x8;

// ---------------------------------------------------------------------------
// WMMA helper (CDNA5 V_WMMA_F32_16X16X32_BF16, wave32)
// ---------------------------------------------------------------------------
__device__ __forceinline__ f32x8 wmma_bf16(bf16x16 a, bf16x16 b, f32x8 c) {
  return __builtin_amdgcn_wmma_f32_16x16x32_bf16(
      /*neg_a=*/false, a, /*neg_b=*/false, b,
      /*c_mod=*/(short)0, c, /*reuse_a=*/false, /*reuse_b=*/false);
}

// A fragment (16M x 32K bf16) from a row-major [M][K] LDS tile.
// lane&15 = M row; lanes 0-15 hold K 0..15, lanes 16-31 hold K 16..31.
// Per lane: 32 contiguous bytes -> 2x ds_load_b128.
__device__ __forceinline__ bf16x16 frag_a_rowmajor(const __bf16* lds, int stride,
                                                   int r0, int k0, int lane) {
  const int half = (lane >> 4) & 1;
  const __bf16* p = lds + (r0 + (lane & 15)) * stride + k0 + half * 16;
  bf16x16 f;
#pragma unroll
  for (int j = 0; j < 16; ++j) f[j] = p[j];
  return f;
}

// B fragment (32K x 16N bf16) from a column-major store: lds[n][k] with k
// contiguous.  lane&15 = N col; lanes 0-15 hold K 0..15, lanes 16-31 K 16..31.
// Per lane: 32 contiguous bytes -> 2x ds_load_b128.
__device__ __forceinline__ bf16x16 frag_b_colmajor(const __bf16* lds, int stride,
                                                   int k0, int c0, int lane) {
  const int half = (lane >> 4) & 1;
  const __bf16* p = lds + (c0 + (lane & 15)) * stride + k0 + half * 16;
  bf16x16 f;
#pragma unroll
  for (int j = 0; j < 16; ++j) f[j] = p[j];
  return f;
}

// ---------------------------------------------------------------------------
// Kernel 1: mean over Q.  One wave per (b,l,n) row of 1024 contiguous floats.
// Writes mean[(b*N + n)*L + l].
// ---------------------------------------------------------------------------
__global__ void k_mean(const float* __restrict__ x, float* __restrict__ mean) {
  const int r    = (blockIdx.x * blockDim.x + threadIdx.x) >> 5; // wave = row
  const int lane = threadIdx.x & 31;
  const int b = r >> 14;
  const int l = (r >> 8) & 63;
  const int n = r & 255;
  const float4* row = (const float4*)(x + (size_t)r * QQ);
  float s = 0.f;
#pragma unroll
  for (int i = 0; i < 8; ++i) {
    float4 v = row[lane + i * 32];
    s += v.x + v.y + v.z + v.w;
  }
#pragma unroll
  for (int off = 16; off > 0; off >>= 1) s += __shfl_down(s, off, 32);
  if (lane == 0) mean[((size_t)(b * NN + n)) * LL + l] = s * (1.0f / (float)QQ);
}

// ---------------------------------------------------------------------------
// threefry2x32 with key (0,42)  (jax.random.key(42)); JAX counter scheme:
// counts = iota(n); halves (i, i+n/2) -> out0, (i-n/2, i) -> out1.
// ---------------------------------------------------------------------------
__device__ __forceinline__ void tf2x32(uint32_t x0, uint32_t x1,
                                       uint32_t& o0, uint32_t& o1) {
  const uint32_t k0 = 0u, k1 = 42u, k2 = 0u ^ 42u ^ 0x1BD11BDAu;
  uint32_t X0 = x0 + k0, X1 = x1 + k1;
#define TF_R(rot) { X0 += X1; X1 = (X1 << (rot)) | (X1 >> (32 - (rot))); X1 ^= X0; }
#define TF_R4A TF_R(13) TF_R(15) TF_R(26) TF_R(6)
#define TF_R4B TF_R(17) TF_R(29) TF_R(16) TF_R(24)
  TF_R4A; X0 += k1; X1 += k2 + 1u;
  TF_R4B; X0 += k2; X1 += k0 + 2u;
  TF_R4A; X0 += k0; X1 += k1 + 3u;
  TF_R4B; X0 += k1; X1 += k2 + 4u;
  TF_R4A; X0 += k2; X1 += k0 + 5u;
#undef TF_R4B
#undef TF_R4A
#undef TF_R
  o0 = X0; o1 = X1;
}

__device__ __forceinline__ float gumbel_at(uint32_t i) {
  const uint32_t half = (uint32_t)(BB * NN * NN); // n_total/2 = 262144
  uint32_t a0, a1, bits;
  if (i < half) { tf2x32(i, i + half, a0, a1); bits = a0; }
  else          { tf2x32(i - half, i, a0, a1); bits = a1; }
  uint32_t fb = (bits >> 9) | 0x3f800000u;        // [1,2)
  float f = __builtin_bit_cast(float, fb) - 1.0f; // [0,1)
  float u = fmaxf(f * (1.0f - 1e-20f) + 1e-20f, 1e-20f);
  return -logf(-logf(u));
}

__device__ __forceinline__ float gelu_exact(float v) {
  return 0.5f * v * (1.0f + erff(v * 0.7071067811865476f));
}

// ---------------------------------------------------------------------------
// Kernel 2: corr -> MLP -> gumbel hard argmax -> self loops -> row normalize.
// One block per (b, n) row; thread = column m.  Emits norm as bf16.
// ---------------------------------------------------------------------------
__global__ void k_adj(const float* __restrict__ mean,
                      const float* __restrict__ w1, const float* __restrict__ b1,
                      const float* __restrict__ w2, const float* __restrict__ b2,
                      const float* __restrict__ w3, const float* __restrict__ b3,
                      __bf16* __restrict__ normbf) {
  __shared__ float meanN[LL];
  __shared__ float red[256];
  const int b = blockIdx.x >> 8;
  const int n = blockIdx.x & 255;
  const int m = threadIdx.x;
  if (m < LL) meanN[m] = mean[((size_t)(b * NN + n)) * LL + m];
  __syncthreads();

  const float* rowM = mean + ((size_t)(b * NN + m)) * LL;
  float corr = 0.f;
#pragma unroll 8
  for (int l = 0; l < LL; ++l) corr = fmaf(meanN[l], rowM[l], corr);

  // MLP 1 -> 16 -> 8 -> 2, exact GELU
  float h1[16];
#pragma unroll
  for (int j = 0; j < 16; ++j) h1[j] = gelu_exact(fmaf(corr, w1[j], b1[j]));
  float h2[8];
#pragma unroll
  for (int k = 0; k < 8; ++k) {
    float a = b2[k];
#pragma unroll
    for (int j = 0; j < 16; ++j) a = fmaf(w2[k * 16 + j], h1[j], a);
    h2[k] = gelu_exact(a);
  }
  float z0 = b3[0], z1 = b3[1];
#pragma unroll
  for (int k = 0; k < 8; ++k) {
    z0 = fmaf(w3[k], h2[k], z0);
    z1 = fmaf(w3[8 + k], h2[k], z1);
  }
  // log_softmax (common shift — does not change the argmax, kept for fidelity)
  const float mx  = fmaxf(z0, z1);
  const float lse = mx + logf(expf(z0 - mx) + expf(z1 - mx));
  z0 -= lse; z1 -= lse;

  const uint32_t e = (uint32_t)((b * NN + n) * NN + m);
  const float g0 = gumbel_at(2u * e);
  const float g1 = gumbel_at(2u * e + 1u);
  float adj = ((z0 + g0) >= (z1 + g1)) ? 1.0f : 0.0f; // argmax, ties -> idx 0
  if (m == n) adj = 1.0f;                              // forced self loop

  red[m] = adj;
  __syncthreads();
#pragma unroll
  for (int s2 = 128; s2 > 0; s2 >>= 1) {
    if (m < s2) red[m] += red[m + s2];
    __syncthreads();
  }
  const float nv = adj / red[0];
  normbf[((size_t)(b * NN + n)) * NN + m] = (__bf16)nv;
}

// ---------------------------------------------------------------------------
// Kernel 3 (stage A): Z[b,o,m,q] = sum_l lin_w[o,l] * x[b,l,m,q]  (bf16 out).
// Computed transposed per tile: D[c,o] = X^T(c x l) * lin_w^T(l x o) so that
// both WMMA fragments are per-lane contiguous in LDS (ds_load_b128 path).
// ---------------------------------------------------------------------------
#define XSTRIDE 72   // ldsXt/ldsW row pitch (bf16): 144 B, 16B-aligned, skewed
__global__ void k_lin(const float* __restrict__ x,
                      const float* __restrict__ lin_w,
                      __bf16* __restrict__ Z) {
  __shared__ __attribute__((aligned(16))) __bf16 ldsW[OUTC * XSTRIDE]; // [o][l]
  __shared__ __attribute__((aligned(16))) __bf16 ldsXt[64 * XSTRIDE];  // [c][l]
  __shared__ __attribute__((aligned(16))) __bf16 ldsC[OUTC * 64];      // [o][c]
  const int tid = threadIdx.x;
  const int b   = blockIdx.x >> 12;            // 4096 column-tiles per batch
  const int c0  = (blockIdx.x & 4095) * 64;

  for (int i = tid; i < OUTC * LL; i += 256)
    ldsW[(i >> 6) * XSTRIDE + (i & 63)] = (__bf16)lin_w[i];
  { // load X tile (64 l x 64 c) coalesced; write transposed [c][l]
    const int l = tid >> 2, cseg = (tid & 3) * 16;
    const float* src = x + ((size_t)(b * LL + l)) * COLS + c0 + cseg;
#pragma unroll
    for (int i = 0; i < 16; i += 4) {
      float4 v = *(const float4*)(src + i);
      ldsXt[(cseg + i) * XSTRIDE + l]     = (__bf16)v.x;
      ldsXt[(cseg + i + 1) * XSTRIDE + l] = (__bf16)v.y;
      ldsXt[(cseg + i + 2) * XSTRIDE + l] = (__bf16)v.z;
      ldsXt[(cseg + i + 3) * XSTRIDE + l] = (__bf16)v.w;
    }
  }
  __syncthreads();

  const int lane = tid & 31, wid = tid >> 5;
  const int ci  = wid & 3;        // shared A tile (c rows)
  const int oiA = wid >> 2;       // B tiles oiA and oiA+2
  f32x8 acc0 = {}, acc1 = {};
#pragma unroll
  for (int kk = 0; kk < 2; ++kk) {
    bf16x16 a  = frag_a_rowmajor(ldsXt, XSTRIDE, ci * 16, kk * 32, lane);
    bf16x16 b0 = frag_b_colmajor(ldsW, XSTRIDE, kk * 32, oiA * 16, lane);
    bf16x16 b1 = frag_b_colmajor(ldsW, XSTRIDE, kk * 32, (oiA + 2) * 16, lane);
    acc0 = wmma_bf16(a, b0, acc0);
    acc1 = wmma_bf16(a, b1, acc1);
  }
  { // D is [c x o]; dump as [o][c]: 16 contiguous bytes per lane per tile
    const int half = lane >> 4;
    __bf16* p0 = ldsC + (oiA * 16 + (lane & 15)) * 64 + ci * 16 + half * 8;
    __bf16* p1 = ldsC + ((oiA + 2) * 16 + (lane & 15)) * 64 + ci * 16 + half * 8;
#pragma unroll
    for (int r = 0; r < 8; ++r) p0[r] = (__bf16)acc0[r];
#pragma unroll
    for (int r = 0; r < 8; ++r) p1[r] = (__bf16)acc1[r];
  }
  __syncthreads();
  {
    const int o = tid >> 2, seg = (tid & 3) * 16;
    __bf16* dst = Z + ((size_t)(b * OUTC + o)) * COLS + c0 + seg;
    const uint4* srcv = (const uint4*)(ldsC + o * 64 + seg);
    ((uint4*)dst)[0] = srcv[0];
    ((uint4*)dst)[1] = srcv[1];
  }
}

// ---------------------------------------------------------------------------
// Kernel 4 (stage B): out[b,o,n,q] = sum_m norm[b,n,m]*Z[b,o,m,q] + lin_b[o].
// Computed transposed per tile: D[q,n] = Z^T(q x m) * norm^T(m x n).  The
// row-major norm tile is exactly column-major for norm^T, so B fragments are
// contiguous; the Z tile is transposed into LDS on the write side.
// Z (128 MB bf16) is expected to be largely L2-resident (192 MB L2).
// ---------------------------------------------------------------------------
#define ASTRIDE 264  // norm tile pitch (bf16): 528 B, 16B-aligned, bank-skewed
#define TSTRIDE 40   // Z^T tile pitch (bf16): 80 B, 16B-aligned, bank-skewed
__global__ void k_conv(const __bf16* __restrict__ Z,
                       const __bf16* __restrict__ normbf,
                       const float* __restrict__ lin_b,
                       float* __restrict__ out) {
  __shared__ __attribute__((aligned(16))) __bf16 ldsA[64 * ASTRIDE]; // norm [n][m]
  __shared__ __attribute__((aligned(16))) __bf16 ldsBt[64 * TSTRIDE];// Z^T [q][m]
  __shared__ __attribute__((aligned(16))) float  ldsCf[64 * 64];     // out [n][q]
  const int tid = threadIdx.x;
  const int gid = blockIdx.x;
  const int qi = gid & 15;
  const int ni = (gid >> 4) & 3;
  const int o  = (gid >> 6) & 63;
  const int b  = gid >> 12;
  const int n0 = ni * 64, q0 = qi * 64;

  { // norm tile -> LDS (rows n0..n0+63, all 256 m), 128-bit loads
    const int nn = tid >> 2, seg = (tid & 3) * 64;
    const uint4* src =
        (const uint4*)(normbf + ((size_t)(b * NN + n0 + nn)) * NN + seg);
    uint4* dst = (uint4*)(ldsA + nn * ASTRIDE + seg);
#pragma unroll
    for (int i = 0; i < 8; ++i) dst[i] = src[i];
  }

  const __bf16* zbase = Z + ((size_t)(b * OUTC + o)) * COLS + q0;
  const int lane = tid & 31, wid = tid >> 5;
  const int qt  = wid & 3;        // shared A tile (q rows)
  const int ntA = wid >> 2;       // B tiles ntA and ntA+2
  f32x8 acc0 = {}, acc1 = {};
#pragma unroll 1
  for (int kk = 0; kk < 8; ++kk) {
    __syncthreads(); // previous compute done before overwriting ldsBt
    { // load 32 m x 64 q tile coalesced; write transposed [q][m]
      const int mm = tid >> 3, qseg = (tid & 7) * 8;
      uint4 raw = *(const uint4*)(zbase + (size_t)(kk * 32 + mm) * QQ + qseg);
      __bf16 v[8];
      __builtin_memcpy(v, &raw, 16);
#pragma unroll
      for (int i = 0; i < 8; ++i) ldsBt[(qseg + i) * TSTRIDE + mm] = v[i];
    }
    if (kk < 7) { // pull next K-tile toward L0 (global_prefetch_b8)
      const int mm = tid >> 3;
      __builtin_prefetch(zbase + (size_t)((kk + 1) * 32 + mm) * QQ, 0, 0);
    }
    __syncthreads();
    bf16x16 a  = frag_a_rowmajor(ldsBt, TSTRIDE, qt * 16, 0, lane);
    bf16x16 b0 = frag_b_colmajor(ldsA, ASTRIDE, kk * 32, ntA * 16, lane);
    bf16x16 b1 = frag_b_colmajor(ldsA, ASTRIDE, kk * 32, (ntA + 2) * 16, lane);
    acc0 = wmma_bf16(a, b0, acc0);
    acc1 = wmma_bf16(a, b1, acc1);
  }

  const float bias = lin_b[o];
  __syncthreads();
  { // D is [q x n]; dump as [n][q]: 32 contiguous bytes per lane per tile
    const int half = lane >> 4;
    float* p0 = ldsCf + (ntA * 16 + (lane & 15)) * 64 + qt * 16 + half * 8;
    float* p1 = ldsCf + ((ntA + 2) * 16 + (lane & 15)) * 64 + qt * 16 + half * 8;
#pragma unroll
    for (int r = 0; r < 8; ++r) p0[r] = acc0[r] + bias;
#pragma unroll
    for (int r = 0; r < 8; ++r) p1[r] = acc1[r] + bias;
  }
  __syncthreads();
  {
    const int nn = tid >> 2, seg = (tid & 3) * 16;
    float* dst = out + (((size_t)(b * OUTC + o)) * NN + n0 + nn) * QQ + q0 + seg;
    const float4* src = (const float4*)(ldsCf + nn * 64 + seg);
#pragma unroll
    for (int i = 0; i < 4; ++i) ((float4*)dst)[i] = src[i];
  }
}

// ---------------------------------------------------------------------------
// Launch
// ---------------------------------------------------------------------------
extern "C" void kernel_launch(void* const* d_in, const int* in_sizes, int n_in,
                              void* d_out, int out_size, void* d_ws, size_t ws_size,
                              hipStream_t stream) {
  const float* x     = (const float*)d_in[0];
  const float* w1    = (const float*)d_in[1];
  const float* b1    = (const float*)d_in[2];
  const float* w2    = (const float*)d_in[3];
  const float* b2    = (const float*)d_in[4];
  const float* w3    = (const float*)d_in[5];
  const float* b3    = (const float*)d_in[6];
  const float* lin_w = (const float*)d_in[7];
  const float* lin_b = (const float*)d_in[8];
  float* out = (float*)d_out;

  char* ws = (char*)d_ws;
  float*  mean   = (float*)ws;                          // 256 KB
  __bf16* normbf = (__bf16*)(ws + (256u << 10));        // 512 KB
  __bf16* Z      = (__bf16*)(ws + (1024u << 10));       // 128 MB (bf16, L2-sized)

  k_mean<<<dim3((BB * LL * NN) / 8), dim3(256), 0, stream>>>(x, mean);
  k_adj <<<dim3(BB * NN),            dim3(256), 0, stream>>>(mean, w1, b1, w2, b2,
                                                             w3, b3, normbf);
  k_lin <<<dim3(BB * (COLS / 64)),   dim3(256), 0, stream>>>(x, lin_w, Z);
  k_conv<<<dim3(BB * OUTC * 4 * 16), dim3(256), 0, stream>>>(Z, normbf, lin_b, out);
}
